// SelfAttention_34763465293902
// MI455X (gfx1250) — compile-verified, matrix operand order checked by reading, not run on hardware
//
#include <hip/hip_runtime.h>
#include <hip/hip_bf16.h>

// ---------------------------------------------------------------------------
// Self-attention for MI455X (gfx1250, wave32, WMMA).
//   B=4, T=2048, E=1024, H=16, HS=64
// All GEMMs via v_wmma_f32_16x16x32_bf16 (bf16 inputs, fp32 accumulate).
// Attention stages K/V tiles into LDS with GLOBAL_LOAD_ASYNC_TO_LDS_B128
// (ASYNCcnt) shared by all 8 waves of a block.
// ---------------------------------------------------------------------------

#define BATCH   4
#define SEQ     2048
#define EMB     1024
#define HEADS   16
#define HS      64
#define MROWS   (BATCH * SEQ)          // 8192

typedef __attribute__((ext_vector_type(16))) __bf16 v16bf;
typedef __attribute__((ext_vector_type(8)))  float  v8f;

__device__ inline __bf16 f2bf(float f) { return (__bf16)f; }

// Reduce across the 16 lanes of a half-wave (C-matrix rows live per-half).
__device__ inline float redsum16(float v) {
    v += __shfl_xor(v, 1, 32);
    v += __shfl_xor(v, 2, 32);
    v += __shfl_xor(v, 4, 32);
    v += __shfl_xor(v, 8, 32);
    return v;
}
__device__ inline float redmax16(float v) {
    v = fmaxf(v, __shfl_xor(v, 1, 32));
    v = fmaxf(v, __shfl_xor(v, 2, 32));
    v = fmaxf(v, __shfl_xor(v, 4, 32));
    v = fmaxf(v, __shfl_xor(v, 8, 32));
    return v;
}

// Load one 16x32 bf16 WMMA fragment (A row-major, or B with column-contiguous
// source). ISA 16-bit A layout (CDNA5 ISA §7.12.2):
//   lanes 0-15 : row = lane,    K in {0..7} U {16..23}
//   lanes 16-31: row = lane-16, K in {8..15} U {24..31}
// Each VGPR holds a contiguous (k, k+1) pair -> one aligned dword load.
// Works for global or LDS-derived pointers (addrspace inferred after inline).
__device__ inline v16bf load_frag16(const __bf16* __restrict__ base, int ld, int lane) {
    const int rc = lane & 15;
    const int hi = (lane >> 4) & 1;
    const __bf16* p = base + (size_t)rc * ld + hi * 8;
    v16bf f;
#pragma unroll
    for (int vg = 0; vg < 8; ++vg) {
        const int k = (vg < 4) ? (2 * vg) : (16 + 2 * (vg - 4));
        f[2 * vg]     = p[k];
        f[2 * vg + 1] = p[k + 1];
    }
    return f;
}

// Async copy 16 bytes global -> LDS (ASYNCcnt-tracked, no VGPR round-trip).
// lds_off is the byte offset from the wave's LDS base (single __shared__
// block => offset 0 for the block start).
__device__ inline void async_copy_b128(unsigned lds_off, const void* gptr) {
    unsigned long long ga = (unsigned long long)gptr;
    asm volatile("global_load_async_to_lds_b128 %0, %1, off"
                 :: "v"(lds_off), "v"(ga) : "memory");
}
__device__ inline void async_wait0() {
    asm volatile("s_wait_asynccnt 0x0" ::: "memory");
}

// ---------------------------------------------------------------------------
// Kernel 1: fp32 -> bf16 conversion
// ---------------------------------------------------------------------------
__global__ __launch_bounds__(256) void cvt_bf16_kernel(const float* __restrict__ src,
                                                       __bf16* __restrict__ dst, int n) {
    int i = blockIdx.x * 256 + threadIdx.x;
    if (i < n) dst[i] = f2bf(src[i]);
}

// ---------------------------------------------------------------------------
// Kernel 2: fused QKV projection + per-head LayerNorm(K,Q) + 1/e^0.25 scale.
// grid = (512, 2), block = 256 (8 waves). Wave tile: 16 rows x 64 cols = one
// head, so LN stats are in-register + 16-lane reductions.
// Outputs: Qb/Kb [b,h,t,d] bf16 ; Vt [b,h,d,t] bf16 (pre-transposed for PV).
// ---------------------------------------------------------------------------
__global__ __launch_bounds__(256) void qkv_ln_kernel(
    const __bf16* __restrict__ xb,                       // [8192,1024]
    const __bf16* __restrict__ wkb, const __bf16* __restrict__ wqb,
    const __bf16* __restrict__ wvb,
    const float* __restrict__ kg, const float* __restrict__ kbv,
    const float* __restrict__ qg, const float* __restrict__ qbv,
    __bf16* __restrict__ Qb, __bf16* __restrict__ Kb, __bf16* __restrict__ Vt)
{
    const int lane = threadIdx.x & 31;
    const int wave = threadIdx.x >> 5;
    const int mtile = blockIdx.x;                  // 0..511
    const int head  = blockIdx.y * 8 + wave;       // 0..15
    const int m0  = mtile * 16;
    const int col = lane & 15;
    const int hi  = (lane >> 4) & 1;

    v8f accK[4], accQ[4], accV[4];
#pragma unroll
    for (int f = 0; f < 4; ++f) { v8f z = {}; accK[f] = z; accQ[f] = z; accV[f] = z; }

    const int nb = head * HS;
    for (int k0 = 0; k0 < EMB; k0 += 32) {
        v16bf a = load_frag16(xb + (size_t)m0 * EMB + k0, EMB, lane);
#pragma unroll
        for (int f = 0; f < 4; ++f) {
            const size_t wo = (size_t)(nb + f * 16) * EMB + k0;
            v16bf bk = load_frag16(wkb + wo, EMB, lane);
            accK[f] = __builtin_amdgcn_wmma_f32_16x16x32_bf16(false, a, false, bk,
                                                              (short)0, accK[f], false, false);
            v16bf bq = load_frag16(wqb + wo, EMB, lane);
            accQ[f] = __builtin_amdgcn_wmma_f32_16x16x32_bf16(false, a, false, bq,
                                                              (short)0, accQ[f], false, false);
            v16bf bv = load_frag16(wvb + wo, EMB, lane);
            accV[f] = __builtin_amdgcn_wmma_f32_16x16x32_bf16(false, a, false, bv,
                                                              (short)0, accV[f], false, false);
        }
    }

    // Epilogue: per-head LN on K,Q; scale by 1/emb^0.25 = 1/sqrt(32); emit bf16.
    const float invscale = 0.17677669529663687f;   // 1024^-0.25
    const int bidx  = m0 / SEQ;
    const int tbase = m0 % SEQ;

#pragma unroll
    for (int vg = 0; vg < 8; ++vg) {
        const int row = vg + 8 * hi;
        const int t   = tbase + row;
        float sK = 0.f, sQ = 0.f, s2K = 0.f, s2Q = 0.f;
#pragma unroll
        for (int f = 0; f < 4; ++f) {
            const float xk = accK[f][vg]; sK += xk; s2K += xk * xk;
            const float xq = accQ[f][vg]; sQ += xq; s2Q += xq * xq;
        }
        sK = redsum16(sK); s2K = redsum16(s2K);
        sQ = redsum16(sQ); s2Q = redsum16(s2Q);
        const float muK = sK * (1.f / 64.f), muQ = sQ * (1.f / 64.f);
        const float rK = rsqrtf(fmaxf(s2K * (1.f / 64.f) - muK * muK, 0.f) + 1e-5f);
        const float rQ = rsqrtf(fmaxf(s2Q * (1.f / 64.f) - muQ * muQ, 0.f) + 1e-5f);

        const size_t obase = ((size_t)(bidx * HEADS + head) * SEQ + t) * HS;
        const size_t vbase = ((size_t)(bidx * HEADS + head) * HS) * SEQ + t;
#pragma unroll
        for (int f = 0; f < 4; ++f) {
            const int d = f * 16 + col;
            const float kv = ((accK[f][vg] - muK) * rK * kg[d] + kbv[d]) * invscale;
            const float qv = ((accQ[f][vg] - muQ) * rQ * qg[d] + qbv[d]) * invscale;
            Kb[obase + d] = f2bf(kv);
            Qb[obase + d] = f2bf(qv);
            Vt[vbase + (size_t)d * SEQ] = f2bf(accV[f][vg]);
        }
    }
}

// ---------------------------------------------------------------------------
// Kernel 3: flash attention. grid = (64, 16), block = 256 (8 waves).
// All 8 waves of a block share one (b,h); K/V 32-key tiles are staged in LDS
// with async b128 copies (ASYNCcnt) and consumed by all waves, cutting L2
// traffic 8x. One wave = 16 query rows with online softmax; P goes through
// LDS to convert C-layout -> A-layout for the P*V WMMA.
// Mask semantics match reference exactly: softmax over valid keys only,
// masked-query rows produce 0, denominator gets +1e-13.
//
// LDS layout (single __shared__ block, base offset 0):
//   [    0, 4096)  sK : K tile  [32 keys][64 d]   bf16
//   [ 4096, 8192)  sV : V tile  [64 d  ][32 keys] bf16
//   [ 8192,16384)  sP : P tiles [8 waves][16 q][32 keys] bf16
// ---------------------------------------------------------------------------
#define SK_OFF 0
#define SV_OFF 4096
#define SP_OFF 8192

__global__ __launch_bounds__(256) void attn_kernel(
    const __bf16* __restrict__ Qb, const __bf16* __restrict__ Kb,
    const __bf16* __restrict__ Vt, const int* __restrict__ mask,
    __bf16* __restrict__ ctx)
{
    __shared__ alignas(16) char smem[16384];
    __bf16* sK = (__bf16*)(smem + SK_OFF);
    __bf16* sV = (__bf16*)(smem + SV_OFF);
    __bf16* sP = (__bf16*)(smem + SP_OFF);

    const int tid  = threadIdx.x;
    const int lane = tid & 31;
    const int wave = tid >> 5;
    const int bh    = blockIdx.x;                  // 0..63
    const int qtile = blockIdx.y * 8 + wave;       // 0..127
    const int bidx = bh >> 4, head = bh & 15;
    const int col = lane & 15;
    const int hi  = (lane >> 4) & 1;
    const int q0  = qtile * 16;

    const __bf16* qbase = Qb + ((size_t)bh * SEQ + q0) * HS;
    const v16bf aq0 = load_frag16(qbase,      HS, lane);
    const v16bf aq1 = load_frag16(qbase + 32, HS, lane);

    // Cooperative-staging source addresses (8 bf16 = 16 B per thread apiece).
    // K tile: element block i covers sK[i*8 .. i*8+8) = Kb[kt + i/8][(i%8)*8 ..]
    const __bf16* ksrc = Kb + ((size_t)bh * SEQ) * HS + (size_t)(tid / 8) * HS + (tid % 8) * 8;
    // V tile: sV[i*8 ..) = Vt row d = i/4, keys (i%4)*8 ..
    const __bf16* vsrc = Vt + ((size_t)bh * HS + tid / 4) * SEQ + (tid % 4) * 8;
    const unsigned ldsK = SK_OFF + (unsigned)tid * 16;
    const unsigned ldsV = SV_OFF + (unsigned)tid * 16;

    float rm[8], rl[8];
    v8f O[4];
#pragma unroll
    for (int vg = 0; vg < 8; ++vg) { rm[vg] = -1e30f; rl[vg] = 0.f; }
#pragma unroll
    for (int f = 0; f < 4; ++f) { v8f z = {}; O[f] = z; }

    for (int kt = 0; kt < SEQ; kt += 32) {
        // Stage K (32x64) and V (64x32) tiles into LDS.
        __syncthreads();                               // prior tile fully consumed
        async_copy_b128(ldsK, ksrc + (size_t)kt * HS); // K rows advance by keys
        async_copy_b128(ldsV, vsrc + kt);              // V cols advance by keys
        async_wait0();
        __syncthreads();                               // tile visible to all waves

        v8f s[2];
        float kmv[2];
#pragma unroll
        for (int j = 0; j < 2; ++j) {
            v16bf b0 = load_frag16(sK + j * 16 * HS,      HS, lane);
            v16bf b1 = load_frag16(sK + j * 16 * HS + 32, HS, lane);
            v8f z = {};
            z = __builtin_amdgcn_wmma_f32_16x16x32_bf16(false, aq0, false, b0,
                                                        (short)0, z, false, false);
            z = __builtin_amdgcn_wmma_f32_16x16x32_bf16(false, aq1, false, b1,
                                                        (short)0, z, false, false);
            s[j] = z;
            kmv[j] = (mask[bidx * SEQ + kt + j * 16 + col] != 0) ? 1.0f : 0.0f;
        }

        // Online softmax update (per C-matrix row = vg + 8*hi).
#pragma unroll
        for (int vg = 0; vg < 8; ++vg) {
            const float sm0 = (kmv[0] > 0.f) ? s[0][vg] : -1e30f;
            const float sm1 = (kmv[1] > 0.f) ? s[1][vg] : -1e30f;
            float mx = redmax16(fmaxf(sm0, sm1));
            const float mnew = fmaxf(rm[vg], mx);
            const float corr = __expf(rm[vg] - mnew);
            const float p0 = kmv[0] * __expf(fminf(s[0][vg] - mnew, 80.f));
            const float p1 = kmv[1] * __expf(fminf(s[1][vg] - mnew, 80.f));
            rl[vg] = rl[vg] * corr + redsum16(p0 + p1);
            rm[vg] = mnew;
#pragma unroll
            for (int f = 0; f < 4; ++f) O[f][vg] *= corr;
            const int row = vg + 8 * hi;
            sP[(wave * 16 + row) * 32 + col]      = f2bf(p0);
            sP[(wave * 16 + row) * 32 + 16 + col] = f2bf(p1);
        }

        // P (16x32, wave-private) back as A fragment; O += P * V.
        v16bf ap = load_frag16(sP + wave * 16 * 32, 32, lane);
#pragma unroll
        for (int f = 0; f < 4; ++f) {
            v16bf bv = load_frag16(sV + f * 16 * 32, 32, lane);
            O[f] = __builtin_amdgcn_wmma_f32_16x16x32_bf16(false, ap, false, bv,
                                                           (short)0, O[f], false, false);
        }
    }

    // Finalize: divide by (l + 1e-13), zero masked-query rows, store ctx bf16.
#pragma unroll
    for (int vg = 0; vg < 8; ++vg) {
        const int row = vg + 8 * hi;
        const int t   = q0 + row;
        const float mq  = (mask[bidx * SEQ + t] != 0) ? 1.f : 0.f;
        const float inv = mq / (rl[vg] + 1e-13f);
        const size_t ob = ((size_t)bidx * SEQ + t) * EMB + head * HS;
#pragma unroll
        for (int f = 0; f < 4; ++f)
            ctx[ob + f * 16 + col] = f2bf(O[f][vg] * inv);
    }
}

// ---------------------------------------------------------------------------
// Kernel 4: output projection ctx @ Wo^T + bo -> fp32. grid = (512, 2).
// ---------------------------------------------------------------------------
__global__ __launch_bounds__(256) void outproj_kernel(
    const __bf16* __restrict__ ctx, const __bf16* __restrict__ wob,
    const float* __restrict__ bo, float* __restrict__ out)
{
    const int lane = threadIdx.x & 31;
    const int wave = threadIdx.x >> 5;
    const int mtile = blockIdx.x;                 // 0..511
    const int ntile = blockIdx.y * 8 + wave;      // 0..15 (64 cols each)
    const int m0 = mtile * 16, n0 = ntile * 64;
    const int col = lane & 15;
    const int hi  = (lane >> 4) & 1;

    v8f acc[4];
#pragma unroll
    for (int f = 0; f < 4; ++f) { v8f z = {}; acc[f] = z; }

    for (int k0 = 0; k0 < EMB; k0 += 32) {
        v16bf a = load_frag16(ctx + (size_t)m0 * EMB + k0, EMB, lane);
#pragma unroll
        for (int f = 0; f < 4; ++f) {
            v16bf b = load_frag16(wob + (size_t)(n0 + f * 16) * EMB + k0, EMB, lane);
            acc[f] = __builtin_amdgcn_wmma_f32_16x16x32_bf16(false, a, false, b,
                                                             (short)0, acc[f], false, false);
        }
    }

#pragma unroll
    for (int vg = 0; vg < 8; ++vg) {
        const int row = vg + 8 * hi;
        const size_t ob = (size_t)(m0 + row) * EMB + n0;
#pragma unroll
        for (int f = 0; f < 4; ++f)
            out[ob + f * 16 + col] = acc[f][vg] + bo[n0 + f * 16 + col];
    }
}

// ---------------------------------------------------------------------------
// Host-side launcher. Workspace layout (bytes), total ~88 MB:
//   xb 16M | wk 2M | wq 2M | wv 2M | wo 2M | Qb 16M | Kb 16M | Vt 16M | ctx 16M
// ---------------------------------------------------------------------------
extern "C" void kernel_launch(void* const* d_in, const int* in_sizes, int n_in,
                              void* d_out, int out_size, void* d_ws, size_t ws_size,
                              hipStream_t stream) {
    const float* x    = (const float*)d_in[0];
    const int*   mask = (const int*)d_in[1];
    const float* Wk   = (const float*)d_in[2];
    const float* Wq   = (const float*)d_in[3];
    const float* Wv   = (const float*)d_in[4];
    const float* Wo   = (const float*)d_in[5];
    const float* bo   = (const float*)d_in[6];
    const float* kg   = (const float*)d_in[7];
    const float* kb   = (const float*)d_in[8];
    const float* qg   = (const float*)d_in[9];
    const float* qb   = (const float*)d_in[10];

    char* ws = (char*)d_ws;
    const size_t XB = (size_t)MROWS * EMB * 2;     // 16 MB
    const size_t WB = (size_t)EMB * EMB * 2;       //  2 MB
    const size_t HB = (size_t)MROWS * EMB * 2;     // 16 MB (bf16 [b,h,t,d])
    __bf16* xb  = (__bf16*)(ws);
    __bf16* wkb = (__bf16*)(ws + XB);
    __bf16* wqb = (__bf16*)(ws + XB + WB);
    __bf16* wvb = (__bf16*)(ws + XB + 2 * WB);
    __bf16* wob = (__bf16*)(ws + XB + 3 * WB);
    __bf16* Qb  = (__bf16*)(ws + XB + 4 * WB);
    __bf16* Kb  = (__bf16*)(ws + XB + 4 * WB + HB);
    __bf16* Vt  = (__bf16*)(ws + XB + 4 * WB + 2 * HB);
    __bf16* ctx = (__bf16*)(ws + XB + 4 * WB + 3 * HB);

    const int nx = MROWS * EMB;        // 8388608
    const int nw = EMB * EMB;          // 1048576
    cvt_bf16_kernel<<<(nx + 255) / 256, 256, 0, stream>>>(x,  xb,  nx);
    cvt_bf16_kernel<<<(nw + 255) / 256, 256, 0, stream>>>(Wk, wkb, nw);
    cvt_bf16_kernel<<<(nw + 255) / 256, 256, 0, stream>>>(Wq, wqb, nw);
    cvt_bf16_kernel<<<(nw + 255) / 256, 256, 0, stream>>>(Wv, wvb, nw);
    cvt_bf16_kernel<<<(nw + 255) / 256, 256, 0, stream>>>(Wo, wob, nw);

    qkv_ln_kernel<<<dim3(MROWS / 16, 2), 256, 0, stream>>>(
        xb, wkb, wqb, wvb, kg, kb, qg, qb, Qb, Kb, Vt);

    attn_kernel<<<dim3(BATCH * HEADS, SEQ / 16 / 8), 256, 0, stream>>>(
        Qb, Kb, Vt, mask, ctx);

    outproj_kernel<<<dim3(MROWS / 16, 2), 256, 0, stream>>>(
        ctx, wob, bo, (float*)d_out);
}
